// GRU_67173288509772
// MI455X (gfx1250) — matrix operand order repaired and checked by reading, not compiled, not run on hardware
//
#include <hip/hip_runtime.h>

// MI455X / gfx1250 GRU: WMMA f16 (f32-acc) GEMMs + batch-parallel recurrent scan.
// S=2048 B=256 I=64 H=128 L=2, output [S,B,1] f32.
//
// Round 3: WGP-scope prefetch, gi double-buffer held as f32 (no b16 packing
// movs on the serial path), and x's f32->f16 conversion fused into the
// layer-0 GEMM A-loads (drops a full 200MB memory pass).

typedef __attribute__((ext_vector_type(16))) _Float16 v16h;
typedef __attribute__((ext_vector_type(8)))  float    v8f;

#define S_LEN 2048
#define BATCH 256
#define HID   128
#define G3    384

union V16H { v16h v; uint4 q[2]; };

__device__ __forceinline__ float sigmoid_f(float x) {
    return 1.0f / (1.0f + __expf(-x));
}
__device__ __forceinline__ float tanh_f(float x) {
    float e = __expf(2.0f * x);
    return (e - 1.0f) / (e + 1.0f);
}

// ---------------------------------------------------------------- f32 -> f16
__global__ void cvt_f16(const float* __restrict__ src, _Float16* __restrict__ dst, int n) {
    int i = blockIdx.x * blockDim.x + threadIdx.x;
    if (i < n) dst[i] = (_Float16)src[i];
}

// ----------------------------------------------------- xg = A * W^T + bias
// A: [M, K] row-major (f32 if AF32, else f16), W: [384, K] f16 row-major,
// out: [M, 384] f16. Block = 256 threads = 8 waves; each wave owns one
// 16-row M tile and loops the 24 N tiles. W lives in LDS with rows padded
// to K+8 halves (272B stride -> conflict-free b128 reads).
template <int K, bool AF32>
__global__ __launch_bounds__(256) void gemm_xg(
    const void*     __restrict__ Av,
    const _Float16* __restrict__ W,
    const float*    __restrict__ bias,
    _Float16*       __restrict__ outp)
{
    constexpr int Kp = K + 8;
    constexpr int NF = K / 32;
    extern __shared__ char smem[];
    _Float16* sW = (_Float16*)smem;            // [384][Kp]

    const int tid  = threadIdx.x;
    const int lane = tid & 31;
    const int wv   = tid >> 5;                 // 0..7
    const int nlo  = lane & 15;
    const int hi   = lane >> 4;                // 0 or 1

    // cooperative copy W -> LDS (u32 granularity)
    constexpr int ROW_U32 = K / 2;
    for (int i = tid; i < 384 * ROW_U32; i += 256) {
        int r = i / ROW_U32, c = i - r * ROW_U32;
        ((unsigned*)(sW + r * Kp))[c] = ((const unsigned*)(W + r * K))[c];
    }
    __syncthreads();

    const size_t rowbase = (size_t)blockIdx.x * 128 + (size_t)wv * 16;

    // A fragments: lane holds row M=nlo, K chunks at 32f + 8*hi and +16
    V16H a[NF];
    if constexpr (AF32) {
        const float* arow = (const float*)Av + (rowbase + nlo) * K;
#pragma unroll
        for (int f = 0; f < NF; ++f) {
            int k0 = 32 * f + hi * 8;
            float4 x0 = *(const float4*)(arow + k0);
            float4 x1 = *(const float4*)(arow + k0 + 4);
            float4 x2 = *(const float4*)(arow + k0 + 16);
            float4 x3 = *(const float4*)(arow + k0 + 20);
            a[f].v[0]  = (_Float16)x0.x;  a[f].v[1]  = (_Float16)x0.y;
            a[f].v[2]  = (_Float16)x0.z;  a[f].v[3]  = (_Float16)x0.w;
            a[f].v[4]  = (_Float16)x1.x;  a[f].v[5]  = (_Float16)x1.y;
            a[f].v[6]  = (_Float16)x1.z;  a[f].v[7]  = (_Float16)x1.w;
            a[f].v[8]  = (_Float16)x2.x;  a[f].v[9]  = (_Float16)x2.y;
            a[f].v[10] = (_Float16)x2.z;  a[f].v[11] = (_Float16)x2.w;
            a[f].v[12] = (_Float16)x3.x;  a[f].v[13] = (_Float16)x3.y;
            a[f].v[14] = (_Float16)x3.z;  a[f].v[15] = (_Float16)x3.w;
        }
    } else {
        const _Float16* arow = (const _Float16*)Av + (rowbase + nlo) * K;
#pragma unroll
        for (int f = 0; f < NF; ++f) {
            int k0 = 32 * f + hi * 8;
            a[f].q[0] = *(const uint4*)(arow + k0);
            a[f].q[1] = *(const uint4*)(arow + k0 + 16);
        }
    }

    for (int nt = 0; nt < 24; ++nt) {
        const int ncol = nt * 16 + nlo;
        v8f c = {};
#pragma unroll
        for (int f = 0; f < NF; ++f) {
            V16H b;
            const _Float16* wrow = sW + ncol * Kp + 32 * f + hi * 8;
            b.q[0] = *(const uint4*)(wrow);
            b.q[1] = *(const uint4*)(wrow + 16);
            c = __builtin_amdgcn_wmma_f32_16x16x32_f16(
                    false, a[f].v, false, b.v, (short)0, c, false, false);
        }
        const float bv = bias[ncol];
#pragma unroll
        for (int r = 0; r < 8; ++r) {
            size_t orow = rowbase + (size_t)(hi * 8 + r);
            outp[orow * G3 + nt * 16 + nlo] = (_Float16)(c[r] + bv);
        }
    }
}

// ------------------------------------------------------- recurrent GRU scan
// Grid = 16 blocks (one 16-row batch tile each; batch lanes are independent,
// so no cross-workgroup sync). Block = 256 threads = 8 waves; wave wv owns
// hidden columns [16wv, 16wv+16). Per step each wave: 3 gates x 4 WMMA with
// all 12 W_hh B-fragments resident in VGPRs; h_prev carried in registers.
// LDS: ping-pong h tile [2][16][136] f16 only (8.7 KB, static).
__global__ __launch_bounds__(256) void gru_scan(
    const _Float16* __restrict__ xg,   // [S, BATCH, 384] f16 (incl. b_ih)
    const _Float16* __restrict__ whh,  // [384, 128] f16
    const float*    __restrict__ bhh,  // [384]
    _Float16*       __restrict__ hseq) // [S, BATCH, 128] f16
{
    __shared__ _Float16 sH[2 * 16 * 136];

    const int tid  = threadIdx.x;
    const int lane = tid & 31;
    const int wv   = tid >> 5;
    const int nlo  = lane & 15;
    const int hi   = lane >> 4;
    const int mt   = blockIdx.x;       // batch tile
    const int roff = hi * 8;
    const int jcol = wv * 16 + nlo;

    for (int i = tid; i < 2 * 16 * 136; i += 256) sH[i] = (_Float16)0.0f;

    // loop-invariant W_hh B-fragments, straight from global, kept in VGPRs.
    // lane holds column N = g*128 + jcol; K chunks = 32f + 8*hi .. +7 and +16 .. +23
    V16H bfr[3][4];
#pragma unroll
    for (int g = 0; g < 3; ++g) {
        const _Float16* wrow = whh + (size_t)(g * HID + jcol) * HID;
#pragma unroll
        for (int f = 0; f < 4; ++f) {
            const _Float16* p = wrow + 32 * f + hi * 8;
            bfr[g][f].q[0] = *(const uint4*)(p);
            bfr[g][f].q[1] = *(const uint4*)(p + 16);
        }
    }

    const float bh_r = bhh[jcol];
    const float bh_z = bhh[HID + jcol];
    const float bh_n = bhh[2 * HID + jcol];

    const size_t bbase = (size_t)mt * 16;

    // register double-buffer of input-gate preactivations, held as f32
    // (one full VGPR each -> no sub-word packing movs on the serial path)
    float gir[8], giz[8], gin[8];
    {
        const _Float16* p = xg + bbase * G3;
#pragma unroll
        for (int r = 0; r < 8; ++r) {
            const _Float16* q = p + (size_t)(roff + r) * G3;
            gir[r] = (float)q[jcol];
            giz[r] = (float)q[HID + jcol];
            gin[r] = (float)q[2 * HID + jcol];
        }
    }

    // h_prev carried in registers: lane's C-layout slot (roff+r, jcol) is the
    // exact element this same lane produced last step.
    float hreg[8];
#pragma unroll
    for (int r = 0; r < 8; ++r) hreg[r] = 0.0f;

    __syncthreads();

    int cur = 0;
    for (int t = 0; t < S_LEN; ++t) {
        float cir[8], ciz[8], cin[8];
#pragma unroll
        for (int r = 0; r < 8; ++r) { cir[r] = gir[r]; ciz[r] = giz[r]; cin[r] = gin[r]; }

        // issue loads for t+1 now; they complete while this step computes
        const int tn = (t + 1 < S_LEN) ? (t + 1) : t;
        const _Float16* pn = xg + ((size_t)tn * BATCH + bbase) * G3;
#pragma unroll
        for (int r = 0; r < 8; ++r) {
            const _Float16* q = pn + (size_t)(roff + r) * G3;
            gir[r] = (float)q[jcol];
            giz[r] = (float)q[HID + jcol];
            gin[r] = (float)q[2 * HID + jcol];
        }
        const int tp = (t + 2 < S_LEN) ? (t + 2) : t;
        __builtin_prefetch(xg + ((size_t)tp * BATCH + bbase + roff) * G3 + jcol, 0, 3);

        // A fragments = previous hidden tile from LDS (written by all waves)
        const _Float16* hrow = sH + cur * (16 * 136) + nlo * 136;
        V16H a[4];
#pragma unroll
        for (int f = 0; f < 4; ++f) {
            int k0 = 32 * f + hi * 8;
            a[f].q[0] = *(const uint4*)(hrow + k0);
            a[f].q[1] = *(const uint4*)(hrow + k0 + 16);
        }

        // 3 independent accumulation chains of 4 WMMA each; B already in regs
        v8f cg[3];
#pragma unroll
        for (int g = 0; g < 3; ++g) {
            v8f c = {};
#pragma unroll
            for (int f = 0; f < 4; ++f) {
                c = __builtin_amdgcn_wmma_f32_16x16x32_f16(
                        false, a[f].v, false, bfr[g][f].v, (short)0, c, false, false);
            }
            cg[g] = c;
        }

        const int nxt = cur ^ 1;
        const size_t orow0 = ((size_t)t * BATCH + bbase + (size_t)roff) * HID;
#pragma unroll
        for (int r = 0; r < 8; ++r) {
            float ghr = cg[0][r] + bh_r;
            float ghz = cg[1][r] + bh_z;
            float ghn = cg[2][r] + bh_n;
            float rr  = sigmoid_f(cir[r] + ghr);
            float zz  = sigmoid_f(ciz[r] + ghz);
            float nn  = tanh_f(cin[r] + rr * ghn);
            float hv  = (1.0f - zz) * nn + zz * hreg[r];
            hreg[r] = hv;
            sH[nxt * (16 * 136) + (roff + r) * 136 + jcol] = (_Float16)hv;
            hseq[orow0 + (size_t)r * HID + jcol] = (_Float16)hv;
        }
        __syncthreads();
        cur = nxt;
    }
}

// ------------------------------------------------ out = h1 . W_lin + b_lin
__global__ __launch_bounds__(256) void linear_out(
    const _Float16* __restrict__ h, const float* __restrict__ wlin,
    const float* __restrict__ blin, float* __restrict__ outp, int n)
{
    int i = blockIdx.x * 256 + threadIdx.x;
    if (i >= n) return;
    const _Float16* p = h + (size_t)i * HID;
    float acc = 0.0f;
#pragma unroll
    for (int j = 0; j < HID; ++j) acc += (float)p[j] * wlin[j];
    outp[i] = acc + blin[0];
}

extern "C" void kernel_launch(void* const* d_in, const int* in_sizes, int n_in,
                              void* d_out, int out_size, void* d_ws, size_t ws_size,
                              hipStream_t stream) {
    (void)in_sizes; (void)n_in; (void)ws_size;
    const float* x    = (const float*)d_in[0];
    const float* Wih0 = (const float*)d_in[1];
    const float* Whh0 = (const float*)d_in[2];
    const float* bih0 = (const float*)d_in[3];
    const float* bhh0 = (const float*)d_in[4];
    const float* Wih1 = (const float*)d_in[5];
    const float* Whh1 = (const float*)d_in[6];
    const float* bih1 = (const float*)d_in[7];
    const float* bhh1 = (const float*)d_in[8];
    const float* Wlin = (const float*)d_in[9];
    const float* blin = (const float*)d_in[10];

    char* ws = (char*)d_ws;
    size_t off = 0;
    auto take = [&](size_t bytes) { size_t o = off; off = (off + bytes + 255) & ~(size_t)255; return o; };

    const size_t M = (size_t)S_LEN * BATCH;                       // 524288
    _Float16* wih0h = (_Float16*)(ws + take(384 * 64  * sizeof(_Float16)));
    _Float16* whh0h = (_Float16*)(ws + take(384 * 128 * sizeof(_Float16)));
    _Float16* wih1h = (_Float16*)(ws + take(384 * 128 * sizeof(_Float16)));
    _Float16* whh1h = (_Float16*)(ws + take(384 * 128 * sizeof(_Float16)));
    _Float16* xgbuf = (_Float16*)(ws + take(M * G3  * sizeof(_Float16)));   // 403 MB (reused both layers)
    _Float16* h0seq = (_Float16*)(ws + take(M * HID * sizeof(_Float16)));   // 134 MB
    _Float16* h1seq = (_Float16*)(ws + take(M * HID * sizeof(_Float16)));   // 134 MB

    // allow >64KB dynamic LDS for the K=128 GEMM
    (void)hipFuncSetAttribute((const void*)gemm_xg<128, false>,
        hipFuncAttributeMaxDynamicSharedMemorySize, 104448);

    // weight f32 -> f16 converts (tiny)
    cvt_f16<<<(384 * 64  + 255) / 256, 256, 0, stream>>>(Wih0, wih0h, 384 * 64);
    cvt_f16<<<(384 * 128 + 255) / 256, 256, 0, stream>>>(Whh0, whh0h, 384 * 128);
    cvt_f16<<<(384 * 128 + 255) / 256, 256, 0, stream>>>(Wih1, wih1h, 384 * 128);
    cvt_f16<<<(384 * 128 + 255) / 256, 256, 0, stream>>>(Whh1, whh1h, 384 * 128);

    const int gemm_blocks = (int)(M / 128);          // 4096
    const size_t shm64  = 384 * (64  + 8) * sizeof(_Float16);   //  55296
    const size_t shm128 = 384 * (128 + 8) * sizeof(_Float16);   // 104448

    // Layer 0 (A = x in f32, converted in-register during fragment load)
    gemm_xg<64, true><<<gemm_blocks, 256, shm64, stream>>>(x, wih0h, bih0, xgbuf);
    gru_scan<<<16, 256, 0, stream>>>(xgbuf, whh0h, bhh0, h0seq);

    // Layer 1
    gemm_xg<128, false><<<gemm_blocks, 256, shm128, stream>>>(h0seq, wih1h, bih1, xgbuf);
    gru_scan<<<16, 256, 0, stream>>>(xgbuf, whh1h, bhh1, h1seq);

    // Output projection
    linear_out<<<(int)((M + 255) / 256), 256, 0, stream>>>(h1seq, Wlin, blin, (float*)d_out, (int)M);
    (void)out_size;
}
